// GCN_45114336477305
// MI455X (gfx1250) — matrix operand order repaired and checked by reading, not compiled, not run on hardware
//
#include <hip/hip_runtime.h>

// ---------------------------------------------------------------------------
// 2-layer GCN for MI455X (gfx1250, wave32).
//  - GEMMs via v_wmma_f32_16x16x32_bf16 (f32 accumulate), W staged whole in
//    LDS pre-swizzled into B-fragment layout (W1 bf16 = 196KB < 320KB/WGP).
//  - Each wave processes MROWS row tiles so every B-fragment LDS load feeds
//    MROWS back-to-back WMMAs (hides ds latency, cuts LDS traffic/WMMA).
//  - B-fragments double-buffered across the nt loop (ds_load in flight while
//    WMMA executes).
//  - Edge aggregation via global_atomic_add_f32; h tensors (51MB) are
//    L2-resident (192MB L2), so gathers/atomics stay on-chip.
//  - dinv[src] folded into GEMM epilogue; self-loop + bias folded into the
//    aggregation-buffer init; ReLU output stored directly as bf16 for GEMM2.
// ---------------------------------------------------------------------------

#define GCN_N_NODES 100000

typedef __attribute__((ext_vector_type(16))) __bf16 v16bf;
typedef __attribute__((ext_vector_type(8)))  __bf16 v8bf;
typedef __attribute__((ext_vector_type(8)))  float  v8f;

__device__ __forceinline__ void atomic_add_f32(float* p, float v) {
  unsafeAtomicAdd(p, v);   // native global_atomic_add_f32
}

// ---- A-fragment loaders: 16x32 bf16, CDNA5 ISA 7.12.2 layout -------------
// lane m = lane&15 holds row m; lanes 0-15: K = kb..kb+7, kb+16..kb+23 with
// kb = kt*32 (kg=0); lanes 16-31: same with kb = kt*32+8 (kg=1).
__device__ __forceinline__ v16bf load_afrag(const float* arow, int kb) {
  const float4* p0 = reinterpret_cast<const float4*>(arow + kb);
  const float4* p1 = reinterpret_cast<const float4*>(arow + kb + 16);
  float4 f0 = p0[0], f1 = p0[1], f2 = p1[0], f3 = p1[1];
  v16bf a;
  a[0] = (__bf16)f0.x; a[1] = (__bf16)f0.y; a[2]  = (__bf16)f0.z; a[3]  = (__bf16)f0.w;
  a[4] = (__bf16)f1.x; a[5] = (__bf16)f1.y; a[6]  = (__bf16)f1.z; a[7]  = (__bf16)f1.w;
  a[8] = (__bf16)f2.x; a[9] = (__bf16)f2.y; a[10] = (__bf16)f2.z; a[11] = (__bf16)f2.w;
  a[12] = (__bf16)f3.x; a[13] = (__bf16)f3.y; a[14] = (__bf16)f3.z; a[15] = (__bf16)f3.w;
  return a;
}
__device__ __forceinline__ v16bf load_afrag(const __bf16* arow, int kb) {
  v8bf lo = *reinterpret_cast<const v8bf*>(arow + kb);
  v8bf hi = *reinterpret_cast<const v8bf*>(arow + kb + 16);
  v16bf a;
#pragma unroll
  for (int i = 0; i < 8; ++i) { a[i] = lo[i]; a[i + 8] = hi[i]; }
  return a;
}

// ---- Row-streaming WMMA GEMM: C[row,:] = (A[row,:] @ W) * rowscale[row] ----
// Block = 256 threads = 8 waves; wave w owns MROWS row tiles per step and all
// NOUT column tiles; W lives fully in LDS, pre-swizzled per-fragment so each
// lane's B-frag is one contiguous 32B LDS read.
template <int K, int NOUT, int MROWS, typename AT>
__global__ __launch_bounds__(256) void gemm_rows_wmma(
    const AT* __restrict__ A, const float* __restrict__ W,
    const float* __restrict__ rowscale, float* __restrict__ C, int M) {
  constexpr int KT = K / 32, NT = NOUT / 16;
  extern __shared__ __bf16 wlds[];  // frag (kt*NT+nt): 32 lanes x 16 elems

  // Cooperative load + f32->bf16 convert + swizzle of W into fragment layout.
  for (int idx = threadIdx.x; idx < K * NOUT; idx += 256) {
    int k = idx / NOUT, n = idx - k * NOUT;
    int kt = k >> 5, kr = k & 31;
    int nt = n >> 4, nl = n & 15;
    int lane2 = (((kr >> 3) & 1) << 4) | nl;   // K-group selects lane half
    int e = (kr & 7) | ((kr >> 4) << 3);
    wlds[((((kt * NT + nt) << 5) + lane2) << 4) + e] = (__bf16)W[idx];
  }
  __syncthreads();

  const int wave = threadIdx.x >> 5;
  const int lane = threadIdx.x & 31;
  const int nl = lane & 15;
  const int kg = lane >> 4;
  const v16bf* bfrag = reinterpret_cast<const v16bf*>(wlds);
  const int rowtiles = M >> 4;  // M = 100000 divisible by 16

  for (int g = blockIdx.x * 8 + wave; g * MROWS < rowtiles; g += gridDim.x * 8) {
    // MROWS row tiles; tail tiles clamp to the last tile (writes identical
    // values from the same wave -> benign + deterministic).
    int rt[MROWS];
    const AT* arow[MROWS];
#pragma unroll
    for (int m = 0; m < MROWS; ++m) {
      int r = g * MROWS + m;
      rt[m] = (r < rowtiles) ? r : (rowtiles - 1);
      arow[m] = A + (size_t)(rt[m] * 16 + nl) * K;
    }
    // speculative prefetch of next row group -> global_prefetch_b8
    __builtin_prefetch(
        A + (size_t)((g + gridDim.x * 8) * MROWS * 16 + nl) * K, 0, 1);

    v8f acc[MROWS][NT];
#pragma unroll
    for (int m = 0; m < MROWS; ++m)
#pragma unroll
      for (int nt = 0; nt < NT; ++nt)
        acc[m][nt] = v8f{0.f, 0.f, 0.f, 0.f, 0.f, 0.f, 0.f, 0.f};

#pragma unroll 2
    for (int kt = 0; kt < KT; ++kt) {
      v16bf a[MROWS];
#pragma unroll
      for (int m = 0; m < MROWS; ++m)
        a[m] = load_afrag(arow[m], kt * 32 + kg * 8);

      // nt loop with double-buffered B fragments (static under full unroll)
      v16bf bq[2];
      bq[0] = bfrag[(kt * NT) * 32 + lane];
#pragma unroll
      for (int nt = 0; nt < NT; ++nt) {
        if (nt + 1 < NT) bq[(nt + 1) & 1] = bfrag[(kt * NT + nt + 1) * 32 + lane];
#pragma unroll
        for (int m = 0; m < MROWS; ++m)
          acc[m][nt] = __builtin_amdgcn_wmma_f32_16x16x32_bf16(
              false, a[m], false, bq[nt & 1], (short)0, acc[m][nt], false, false);
      }
    }

    // C layout: VGPR r -> M = kg*8 + r, N = lane&15 (ISA 7.12.2)
#pragma unroll
    for (int m = 0; m < MROWS; ++m) {
#pragma unroll
      for (int r = 0; r < 8; ++r) {
        int row = rt[m] * 16 + kg * 8 + r;
        float sc = rowscale[row];
        float* crow = C + (size_t)row * NOUT;
#pragma unroll
        for (int nt = 0; nt < NT; ++nt) crow[nt * 16 + nl] = acc[m][nt][r] * sc;
      }
    }
  }
}

// ---- degree / norm kernels -------------------------------------------------
__global__ void deg_init_k(float* deg, int n) {
  int i = blockIdx.x * 256 + threadIdx.x;
  if (i < n) deg[i] = 1.0f;  // self loop
}
__global__ void deg_accum_k(const long long* __restrict__ dst, float* deg, int e) {
  int i = blockIdx.x * 256 + threadIdx.x;
  if (i < e) atomic_add_f32(&deg[(int)dst[i]], 1.0f);
}
__global__ void deg_rsqrt_k(float* deg, int n) {
  int i = blockIdx.x * 256 + threadIdx.x;
  if (i < n) deg[i] = rsqrtf(deg[i]);
}

// ---- agg init: agg[i,c] = hs[i,c]*dinv[i] + bias[c] (self loop + bias) -----
template <int F>
__global__ void agg_init_k(const float* __restrict__ hs,
                           const float* __restrict__ dinv,
                           const float* __restrict__ bias,
                           float* __restrict__ agg, int total) {
  int idx = blockIdx.x * 256 + threadIdx.x;
  if (idx >= total) return;
  int i = idx / F, c = idx - i * F;
  agg[idx] = hs[idx] * dinv[i] + bias[c];
}

// ---- edge scatter: agg[dst,:] += hs[src,:] * dinv[dst] ---------------------
template <int F>
__global__ __launch_bounds__(256) void scatter_k(
    const long long* __restrict__ src, const long long* __restrict__ dst,
    const float* __restrict__ hs, const float* __restrict__ dinv,
    float* __restrict__ agg, int E) {
  constexpr int TPE = F / 4;       // threads per edge (float4 each)
  constexpr int EPB = 256 / TPE;   // edges per block
  int e = blockIdx.x * EPB + threadIdx.x / TPE;
  if (e >= E) return;
  int c = (threadIdx.x % TPE) * 4;
  int s = (int)src[e], d = (int)dst[e];
  float w = dinv[d];  // dinv[src] already folded into hs
  float4 v = *reinterpret_cast<const float4*>(hs + (size_t)s * F + c);
  float* out = agg + (size_t)d * F + c;
  atomic_add_f32(out + 0, v.x * w);
  atomic_add_f32(out + 1, v.y * w);
  atomic_add_f32(out + 2, v.z * w);
  atomic_add_f32(out + 3, v.w * w);
}

// ---- ReLU -> bf16 (only consumer is GEMM2's A operand) ---------------------
__global__ void relu_bf16_k(const float* __restrict__ a, __bf16* __restrict__ z,
                            int n) {
  int i = blockIdx.x * 256 + threadIdx.x;
  if (i < n) z[i] = (__bf16)fmaxf(a[i], 0.0f);
}

// ---------------------------------------------------------------------------
extern "C" void kernel_launch(void* const* d_in, const int* in_sizes, int n_in,
                              void* d_out, int out_size, void* d_ws,
                              size_t ws_size, hipStream_t stream) {
  const float* x      = (const float*)d_in[0];
  const long long* ei = (const long long*)d_in[1];  // int64 per reference
  const float* W1     = (const float*)d_in[2];
  const float* b1     = (const float*)d_in[3];
  const float* W2     = (const float*)d_in[4];
  const float* b2     = (const float*)d_in[5];
  float* out = (float*)d_out;

  const int N = GCN_N_NODES;
  const int E = in_sizes[1] / 2;
  const long long* srcp = ei;
  const long long* dstp = ei + E;

  // workspace layout (~129 MB)
  char* ws = (char*)d_ws;
  size_t off = 0;
  auto take = [&](size_t bytes) -> void* {
    void* p = ws + off;
    off += (bytes + 255) & ~(size_t)255;
    return p;
  };
  float*  dinv = (float*)take((size_t)N * 4);           // deg -> dinv in place
  float*  h1s  = (float*)take((size_t)N * 128 * 4);     // (x@W1)*dinv[row]
  float*  agg1 = (float*)take((size_t)N * 128 * 4);
  __bf16* z1   = (__bf16*)take((size_t)N * 128 * 2);    // relu(agg1) bf16
  float*  h2s  = h1s;                                   // reuse (h1s dead)

  // 1) symmetric-norm degrees (self loops included)
  deg_init_k<<<(N + 255) / 256, 256, 0, stream>>>(dinv, N);
  deg_accum_k<<<(E + 255) / 256, 256, 0, stream>>>(dstp, dinv, E);
  deg_rsqrt_k<<<(N + 255) / 256, 256, 0, stream>>>(dinv, N);

  // 2) layer 1 GEMM: h1s = (x @ W1) * dinv[row]   (W1 bf16 in LDS: 196608 B)
  gemm_rows_wmma<768, 128, 2, float>
      <<<256, 256, 768 * 128 * 2, stream>>>(x, W1, dinv, h1s, N);

  // 3) agg1 = self-loop + bias; then scatter edges; then ReLU -> bf16
  agg_init_k<128><<<(N * 128 + 255) / 256, 256, 0, stream>>>(h1s, dinv, b1,
                                                             agg1, N * 128);
  scatter_k<128><<<(E + 7) / 8, 256, 0, stream>>>(srcp, dstp, h1s, dinv, agg1, E);
  relu_bf16_k<<<(N * 128 + 255) / 256, 256, 0, stream>>>(agg1, z1, N * 128);

  // 4) layer 2 GEMM: h2s = (relu @ W2) * dinv[row]  (W2 bf16 in LDS: 16384 B)
  gemm_rows_wmma<128, 64, 4, __bf16>
      <<<256, 256, 128 * 64 * 2, stream>>>(z1, W2, dinv, h2s, N);

  // 5) out = self-loop + bias; scatter edges (no ReLU on output)
  agg_init_k<64><<<(N * 64 + 255) / 256, 256, 0, stream>>>(h2s, dinv, b2, out,
                                                           N * 64);
  scatter_k<64><<<(E + 15) / 16, 256, 0, stream>>>(srcp, dstp, h2s, dinv, out, E);
}
